// Model_Ner_82128364634587
// MI455X (gfx1250) — matrix operand order repaired and checked by reading, not compiled, not run on hardware
//
#include <hip/hip_runtime.h>

#define B_   512
#define T_   128
#define E_   128
#define H_   256
#define L_   10
#define KTOT (E_ + H_)   // 384
#define G4H  (4 * H_)    // 1024

typedef __attribute__((ext_vector_type(8)))  __bf16 v8bf;
typedef __attribute__((ext_vector_type(16))) __bf16 v16bf;
typedef __attribute__((ext_vector_type(8)))  float  v8f;
typedef __attribute__((ext_vector_type(4)))  int    v4i;

typedef __attribute__((address_space(1))) v4i* gv4i_p;   // global int4*
typedef __attribute__((address_space(3))) v4i* lv4i_p;   // LDS int4*

#if __has_builtin(__builtin_amdgcn_global_load_async_to_lds_b128)
#define HAVE_ASYNC_LDS 1
#endif

// ---------------- init: zero h ping buffer, c state, loss ----------------
__global__ void init_kernel(__bf16* hping, float* c_st, float* loss_out) {
    int i = blockIdx.x * blockDim.x + threadIdx.x;   // 0 .. 2*B*H-1
    hping[i] = (__bf16)0.0f;
    c_st[i]  = 0.0f;
    if (i == 0) loss_out[0] = 0.0f;
}

// ---------------- embedding gather -> bf16 ----------------
__global__ void embed_kernel(const int* __restrict__ tokens,
                             const float* __restrict__ emb,
                             __bf16* __restrict__ xb) {
    const int bt = blockIdx.x;          // 0..B*T-1
    const int e  = threadIdx.x;         // 0..E-1
    const int tok = tokens[bt];
    xb[(size_t)bt * E_ + e] = (__bf16)emb[(size_t)tok * E_ + e];
}

// ---------------- sequence lengths ----------------
__global__ void lens_kernel(const int* __restrict__ tokens, int* __restrict__ lens) {
    __shared__ int s[T_];
    const int b = blockIdx.x, t = threadIdx.x;
    s[t] = (tokens[(size_t)b * T_ + t] != 0) ? 1 : 0;
    __syncthreads();
    for (int off = T_ / 2; off > 0; off >>= 1) {
        if (t < off) s[t] += s[t + off];
        __syncthreads();
    }
    if (t == 0) lens[b] = s[0];
}

// ---- pack [k; r] stacked & transposed -> Bc[n][k], n in [0,4H), k in [0,384) ----
__global__ void pack_weights_kernel(const float* __restrict__ wk,   // [E, 4H]
                                    const float* __restrict__ wr,   // [H, 4H]
                                    __bf16* __restrict__ Bc) {      // [4H][KTOT]
    const int idx = blockIdx.x * blockDim.x + threadIdx.x;          // 4H*KTOT
    const int n = idx / KTOT, k = idx % KTOT;
    const float v = (k < E_) ? wk[(size_t)k * G4H + n]
                             : wr[(size_t)(k - E_) * G4H + n];
    Bc[idx] = (__bf16)v;
}

// ---------------- fused LSTM step: z = [x_t | h] @ Bc^T, gates, h/c update ----------------
// Phase 1: async-stage the 16x384 A tile (x_t ‖ h_{t-1}) into LDS (shared by all 8 waves).
// Phase 2: K-loop of bf16 WMMAs, A from LDS, B (packed weights) from L2.
// Phase 3: z tiles -> LDS (overlapping union), gate nonlinearities, h/c update.
union StepSmem {
    __bf16 as[16][KTOT];     // 12 KB, live during phase 1-2
    float  zs[16][G4H];      // 64 KB, live during phase 3
};

__global__ __launch_bounds__(256)
void lstm_step_kernel(const __bf16* __restrict__ xb,     // [B,T,E]
                      const __bf16* __restrict__ Bc,     // [2][4H][KTOT]
                      const __bf16* __restrict__ h_in,   // [2][B,H]
                      __bf16*       __restrict__ h_out,  // [2][B,H]
                      float*        __restrict__ c_st,   // [2][B,H]
                      const float*  __restrict__ bias_f, // [4H]
                      const float*  __restrict__ bias_b, // [4H]
                      __bf16*       __restrict__ hseq,   // [2][B,T,H]
                      int t) {
    __shared__ StepSmem sm;

    const int dir    = blockIdx.x >> 5;                  // 0 fwd, 1 bwd
    const int row0   = (blockIdx.x & 31) * 16;           // batch tile
    const int tt     = dir ? (T_ - 1 - t) : t;
    const int lane   = threadIdx.x & 31;
    const int wave   = threadIdx.x >> 5;                 // 0..7
    const int laneHi = lane >> 4;
    const int lane15 = lane & 15;
    const int n0     = wave * 128;                       // 8 N-tiles per wave

    // ---- phase 1: stage A tile into LDS (16 rows x 768 bytes = 48 16B chunks/row) ----
    for (int c = threadIdx.x; c < 16 * 48; c += 256) {
        const int row = c / 48, ch = c % 48;
        const int b   = row0 + row;
        const __bf16* src = (ch < 16)
            ? xb   + ((size_t)b * T_ + tt) * E_ + ch * 8
            : h_in + ((size_t)dir * B_ + b) * H_ + (ch - 16) * 8;
        __bf16* dst = &sm.as[row][ch * 8];
#ifdef HAVE_ASYNC_LDS
        __builtin_amdgcn_global_load_async_to_lds_b128(
            (gv4i_p)(v4i*)(void*)const_cast<__bf16*>(src),
            (lv4i_p)(v4i*)(void*)dst, 0, 0);
#else
        *(v8bf*)dst = *(const v8bf*)src;
#endif
    }
#ifdef HAVE_ASYNC_LDS
#if __has_builtin(__builtin_amdgcn_s_wait_asynccnt)
    __builtin_amdgcn_s_wait_asynccnt(0);
#else
    asm volatile("s_wait_asynccnt 0" ::: "memory");
#endif
#endif
    __syncthreads();

    // ---- phase 2: GEMM ----
    v8f acc[8] = {};
    const __bf16* bbase = Bc + (size_t)dir * G4H * KTOT;

    #pragma unroll
    for (int kb = 0; kb < KTOT / 32; ++kb) {
        const int kbase = kb * 32;
        // A operand, contiguous k order used here: lanes 0-15 hold K 0..15,
        // lanes 16-31 hold K 16..31 of rows M=0..15 (matches B).
        const int c0 = kbase + (laneHi ? 16 : 0);
        const v8bf alo = *(const v8bf*)&sm.as[lane15][c0];
        const v8bf ahi = *(const v8bf*)&sm.as[lane15][c0 + 8];
        v16bf a;
        #pragma unroll
        for (int i = 0; i < 8; ++i) { a[i] = alo[i]; a[8 + i] = ahi[i]; }

        #pragma unroll
        for (int nt = 0; nt < 8; ++nt) {
            // B operand, same k order as A: lane = column N; hi-lanes hold K 16..31.
            const int n = n0 + nt * 16 + lane15;
            const v16bf bfr = *(const v16bf*)(bbase + (size_t)n * KTOT + kbase
                                              + (laneHi ? 16 : 0));
            acc[nt] = __builtin_amdgcn_wmma_f32_16x16x32_bf16(
                false, a, false, bfr, (short)0, acc[nt], false, false);
        }
    }
    __syncthreads();   // all waves done reading sm.as before zs overlays it

    // ---- phase 3: spill z, gates ----
    // C/D layout: lanes 0-15 -> M = r, lanes 16-31 -> M = 8 + r; N = lane & 15.
    #pragma unroll
    for (int nt = 0; nt < 8; ++nt) {
        const int n = n0 + nt * 16 + lane15;
        #pragma unroll
        for (int r = 0; r < 8; ++r)
            sm.zs[laneHi ? (8 + r) : r][n] = acc[nt][r];
    }
    __syncthreads();

    // prefetch next timestep's x rows while gate math runs
    if (t + 1 < T_ && threadIdx.x < 16) {
        const int ttn = dir ? (tt - 1) : (tt + 1);
        __builtin_prefetch(xb + ((size_t)(row0 + threadIdx.x) * T_ + ttn) * E_, 0, 1);
    }

    const float* bias = dir ? bias_b : bias_f;
    for (int idx = threadIdx.x; idx < 16 * H_; idx += 256) {
        const int m = idx >> 8;                          // row in tile
        const int j = idx & (H_ - 1);                    // hidden index
        const int b = row0 + m;
        const float zi = sm.zs[m][j]          + bias[j];
        const float zf = sm.zs[m][j + H_]     + bias[j + H_];
        const float zg = sm.zs[m][j + 2 * H_] + bias[j + 2 * H_];
        const float zo = sm.zs[m][j + 3 * H_] + bias[j + 3 * H_];
        const size_t cidx = ((size_t)dir * B_ + b) * H_ + j;
        const float cold = c_st[cidx];
        const float ig = 1.0f / (1.0f + __expf(-zi));
        const float fg = 1.0f / (1.0f + __expf(-zf));
        const float og = 1.0f / (1.0f + __expf(-zo));
        const float cn = fg * cold + ig * tanhf(zg);
        const float hn = og * tanhf(cn);
        c_st[cidx]  = cn;
        h_out[cidx] = (__bf16)hn;
        hseq[(((size_t)dir * B_ + b) * T_ + tt) * H_ + j] = (__bf16)hn;
    }
}

// ---------------- emissions = [hf | hb] @ wd ----------------
__global__ void emissions_kernel(const __bf16* __restrict__ hseq,  // [2][B,T,H]
                                 const float* __restrict__ wd,     // [2H, L]
                                 float* __restrict__ em) {         // [B,T,L]
    const int idx = blockIdx.x * blockDim.x + threadIdx.x;
    if (idx >= B_ * T_ * L_) return;
    const int l  = idx % L_;
    const int bt = idx / L_;
    const __bf16* hf = hseq + (size_t)bt * H_;
    const __bf16* hb = hseq + (size_t)B_ * T_ * H_ + (size_t)bt * H_;
    float s = 0.0f;
    for (int j = 0; j < H_; ++j) s += (float)hf[j] * wd[(size_t)j * L_ + l];
    for (int j = 0; j < H_; ++j) s += (float)hb[j] * wd[(size_t)(H_ + j) * L_ + l];
    em[idx] = s;
}

// ---------------- CRF: log-norm + sequence score -> loss ----------------
__global__ void crf_kernel(const float* __restrict__ em,
                           const int* __restrict__ labels,
                           const float* __restrict__ trans,
                           const int* __restrict__ lens,
                           float* __restrict__ loss_out) {
    const int b = blockIdx.x * blockDim.x + threadIdx.x;
    if (b >= B_) return;
    const float* e  = em + (size_t)b * T_ * L_;
    const int len   = lens[b];
    float alpha[L_];
    for (int j = 0; j < L_; ++j) alpha[j] = e[j];
    for (int t = 1; t < T_; ++t) {
        if (t < len) {
            float na[L_];
            for (int j = 0; j < L_; ++j) {
                float mx = -3.0e38f;
                for (int i = 0; i < L_; ++i)
                    mx = fmaxf(mx, alpha[i] + trans[i * L_ + j]);
                float s = 0.0f;
                for (int i = 0; i < L_; ++i)
                    s += __expf(alpha[i] + trans[i * L_ + j] - mx);
                na[j] = mx + __logf(s) + e[t * L_ + j];
            }
            for (int j = 0; j < L_; ++j) alpha[j] = na[j];
        }
    }
    float mx = alpha[0];
    for (int j = 1; j < L_; ++j) mx = fmaxf(mx, alpha[j]);
    float s = 0.0f;
    for (int j = 0; j < L_; ++j) s += __expf(alpha[j] - mx);
    const float lognorm = mx + __logf(s);

    const int* tg = labels + (size_t)b * T_;
    float sc = 0.0f;
    for (int t = 0; t < T_; ++t) if (t < len) sc += e[t * L_ + tg[t]];
    for (int t = 1; t < T_; ++t) if (t < len) sc += trans[tg[t - 1] * L_ + tg[t]];

    atomicAdd(loss_out, -(sc - lognorm));
}

// ---------------- Viterbi decode ----------------
__global__ void viterbi_kernel(const float* __restrict__ em,
                               const float* __restrict__ trans,
                               const int* __restrict__ lens,
                               int* __restrict__ bp_ws,    // [B][T][L]
                               float* __restrict__ tags) { // [B][T]
    const int b = blockIdx.x * blockDim.x + threadIdx.x;
    if (b >= B_) return;
    const float* e = em + (size_t)b * T_ * L_;
    const int len  = lens[b];
    int* bp = bp_ws + (size_t)b * T_ * L_;
    float alpha[L_];
    for (int j = 0; j < L_; ++j) alpha[j] = e[j];
    for (int t = 1; t < T_; ++t) {
        int* bpt = bp + t * L_;
        if (t < len) {
            float na[L_]; int nb[L_];
            for (int j = 0; j < L_; ++j) {
                float best = alpha[0] + trans[j];
                int   bi   = 0;
                for (int i = 1; i < L_; ++i) {
                    const float v = alpha[i] + trans[i * L_ + j];
                    if (v > best) { best = v; bi = i; }
                }
                na[j] = best + e[t * L_ + j];
                nb[j] = bi;
            }
            for (int j = 0; j < L_; ++j) { alpha[j] = na[j]; bpt[j] = nb[j]; }
        } else {
            for (int j = 0; j < L_; ++j) bpt[j] = j;
        }
    }
    int last = 0; float bm = alpha[0];
    for (int j = 1; j < L_; ++j) if (alpha[j] > bm) { bm = alpha[j]; last = j; }
    tags[(size_t)b * T_ + (T_ - 1)] = (float)last;
    int cur = last;
    for (int t = T_ - 1; t >= 1; --t) {
        cur = bp[t * L_ + cur];
        tags[(size_t)b * T_ + (t - 1)] = (float)cur;
    }
}

extern "C" void kernel_launch(void* const* d_in, const int* in_sizes, int n_in,
                              void* d_out, int out_size, void* d_ws, size_t ws_size,
                              hipStream_t stream) {
    const int*   tokens = (const int*)d_in[0];
    const int*   labels = (const int*)d_in[1];
    const float* emb    = (const float*)d_in[2];
    const float* wf_k   = (const float*)d_in[3];
    const float* wf_r   = (const float*)d_in[4];
    const float* bf     = (const float*)d_in[5];
    const float* wb_k   = (const float*)d_in[6];
    const float* wb_r   = (const float*)d_in[7];
    const float* bb     = (const float*)d_in[8];
    const float* wd     = (const float*)d_in[9];
    const float* trans  = (const float*)d_in[10];
    float* out = (float*)d_out;           // out[0] = loss, out[1..] = tags

    char* ws = (char*)d_ws;
    size_t off = 0;
    auto take = [&](size_t bytes) { char* p = ws + off; off = (off + bytes + 255) & ~(size_t)255; return p; };
    __bf16* xb    = (__bf16*)take((size_t)B_ * T_ * E_ * 2);       // 16.8 MB
    __bf16* Bc    = (__bf16*)take((size_t)2 * G4H * KTOT * 2);     // 1.6 MB
    __bf16* hping = (__bf16*)take((size_t)2 * B_ * H_ * 2);
    __bf16* hpong = (__bf16*)take((size_t)2 * B_ * H_ * 2);
    float*  c_st  = (float*)take((size_t)2 * B_ * H_ * 4);
    __bf16* hseq  = (__bf16*)take((size_t)2 * B_ * T_ * H_ * 2);   // 67 MB
    float*  em    = (float*)take((size_t)B_ * T_ * L_ * 4);
    int*    lens  = (int*)take((size_t)B_ * 4);
    int*    bp    = (int*)take((size_t)B_ * T_ * L_ * 4);
    (void)in_sizes; (void)n_in; (void)out_size; (void)ws_size;

    init_kernel<<<(2 * B_ * H_) / 256, 256, 0, stream>>>(hping, c_st, out);
    embed_kernel<<<B_ * T_, E_, 0, stream>>>(tokens, emb, xb);
    lens_kernel<<<B_, T_, 0, stream>>>(tokens, lens);
    pack_weights_kernel<<<(G4H * KTOT) / 256, 256, 0, stream>>>(wf_k, wf_r, Bc);
    pack_weights_kernel<<<(G4H * KTOT) / 256, 256, 0, stream>>>(wb_k, wb_r,
                                                                Bc + (size_t)G4H * KTOT);

    for (int t = 0; t < T_; ++t) {
        const __bf16* hin  = (t & 1) ? hpong : hping;
        __bf16*       hout = (t & 1) ? hping : hpong;
        lstm_step_kernel<<<64, 256, 0, stream>>>(xb, Bc, hin, hout, c_st,
                                                 bf, bb, hseq, t);
    }

    emissions_kernel<<<(B_ * T_ * L_ + 255) / 256, 256, 0, stream>>>(hseq, wd, em);
    crf_kernel<<<(B_ + 255) / 256, 256, 0, stream>>>(em, labels, trans, lens, out);
    viterbi_kernel<<<(B_ + 255) / 256, 256, 0, stream>>>(em, trans, lens, bp, out + 1);
}
